// MoERouter_7267084665016
// MI455X (gfx1250) — matrix operand order repaired and checked by reading, not compiled, not run on hardware
//
#include <hip/hip_runtime.h>
#include <cstddef>
#include <cstdint>

typedef __bf16 bf16_t;
typedef __attribute__((ext_vector_type(16))) __bf16    v16bf;
typedef __attribute__((ext_vector_type(8)))  float     v8f;
typedef __attribute__((ext_vector_type(4)))  float     v4f;
typedef __attribute__((ext_vector_type(4)))  uint32_t  u32x4;
typedef __attribute__((ext_vector_type(8)))  uint32_t  u32x8;

#define H_DIM 4096
#define N_EXP 64
#define N_TOK 16384        // B*S = 4*4096
#define KP    128          // K-panel per TDM stage
#define NPAN  (H_DIM / KP) // 32 panels

// ---------------------------------------------------------------------------
// Pre-pass: split fp32 gate weight into bf16 hi/lo (3xBF16 fp32 emulation).
// ---------------------------------------------------------------------------
__global__ void gate_split_kernel(const float* __restrict__ g,
                                  bf16_t* __restrict__ ghi,
                                  bf16_t* __restrict__ glo) {
    int i = blockIdx.x * blockDim.x + threadIdx.x;   // over E*H = 262144
    float a  = g[i];
    bf16_t h = (bf16_t)a;
    ghi[i]   = h;
    glo[i]   = (bf16_t)(a - (float)h);
}

__device__ __forceinline__ uint32_t rfl(uint32_t x) {
    return __builtin_amdgcn_readfirstlane(x);
}

// Issue one TDM load: 2-D tensor [N_EXP rows x H_DIM], tile [64 x KP], 2-byte
// elements, row-major into LDS at lds_off. Descriptor layout per D# groups 0/1.
__device__ __forceinline__ void tdm_load_panel(const bf16_t* gsrc, uint32_t lds_off) {
    uint64_t ga = (uint64_t)(uintptr_t)gsrc;
    u32x4 g0;
    g0[0] = rfl(1u);                                                  // count=1
    g0[1] = rfl(lds_off);                                             // lds_addr
    g0[2] = rfl((uint32_t)ga);                                        // global_addr lo
    g0[3] = rfl((uint32_t)((ga >> 32) & 0x1FFFFFFu) | (2u << 30));    // ga hi | type=2
    u32x8 g1;
    g1[0] = rfl(1u << 16);                                            // data_size=2B
    g1[1] = rfl(((uint32_t)H_DIM & 0xFFFFu) << 16);                   // tensor_dim0 lo16
    g1[2] = rfl((((uint32_t)H_DIM >> 16) & 0xFFFFu)                   // tensor_dim0 hi16
              | (((uint32_t)N_EXP & 0xFFFFu) << 16));                 // tensor_dim1 lo16
    g1[3] = rfl(((uint32_t)KP & 0xFFFFu) << 16);                      // tile_dim0=KP
    g1[4] = rfl((uint32_t)N_EXP);                                     // tile_dim1=64
    g1[5] = rfl((uint32_t)H_DIM);                                     // dim0_stride lo32
    g1[6] = rfl(0u);
    g1[7] = rfl(0u);
    asm volatile("tensor_load_to_lds %0, %1" : : "s"(g0), "s"(g1) : "memory");
}

__device__ __forceinline__ void top2_insert(float v, int id,
                                            float& m1, int& i1,
                                            float& m2, int& i2) {
    if (v > m1 || (v == m1 && id < i1)) {
        m2 = m1; i2 = i1; m1 = v; i1 = id;
    } else if (v > m2 || (v == m2 && id < i2)) {
        m2 = v; i2 = id;
    }
}

// ---------------------------------------------------------------------------
// Fused router: one wave32 = 16 tokens x 64 experts.
//  - gate bf16 hi/lo panels staged to LDS by the Tensor Data Mover (double buffer)
//  - hidden streamed via register-double-buffered global_load_b128
//  - B LDS operands software-pipelined one K-chunk ahead (double buffer)
//  - 12 v_wmma_f32_16x16x32_bf16 per K=32 chunk (3xBF16 fp32 emulation, 4 N-tiles)
//  - epilogue: logits store, 16-lane butterfly top-2, 2-way softmax weights
// ---------------------------------------------------------------------------
__global__ __launch_bounds__(256, 1) void moe_router_kernel(
    const float*  __restrict__ hidden,
    const bf16_t* __restrict__ ghi,
    const bf16_t* __restrict__ glo,
    float* __restrict__ logits_out,
    float* __restrict__ w_out,
    float* __restrict__ idx_out)
{
    __shared__ bf16_t sB[2][2][N_EXP * KP];   // [stage][hi/lo][e*KP + k]  = 64 KB

    const int tid  = threadIdx.x;
    const int lane = tid & 31;
    const int wv   = tid >> 5;
    const int wave = blockIdx.x * 8 + wv;
    const int lsub = lane & 15;
    const int half = lane >> 4;
    const int t0   = wave * 16;

    // A operand (contiguous k order, same as B): lanes 0-15 -> K {0..15},
    // lanes 16-31 -> K {16..31}; row M = lane%16.
    const float* arow = hidden + (size_t)(t0 + lsub) * H_DIM + half * 16;

    const uint32_t ldsb = (uint32_t)(uintptr_t)(&sB[0][0][0]);
    const uint32_t panel_bytes = (uint32_t)(N_EXP * KP * 2);

    // prologue: stage panel 0 (hi+lo)
    if (wv == 0) {
        tdm_load_panel(ghi, ldsb);
        tdm_load_panel(glo, ldsb + panel_bytes);
    }

    v8f acc[4] = {v8f{}, v8f{}, v8f{}, v8f{}};
    v4f a[2][4];
    a[0][0] = *(const v4f*)(arow + 0);
    a[0][1] = *(const v4f*)(arow + 4);
    a[0][2] = *(const v4f*)(arow + 8);
    a[0][3] = *(const v4f*)(arow + 12);

    for (int p = 0; p < NPAN; ++p) {
        if (wv == 0) {
            if (p + 1 < NPAN) {
                // queue next panel into the other LDS stage, then wait for
                // the current panel's two TDM ops (in-order per wave)
                const uint32_t soff = ldsb + (uint32_t)(((p + 1) & 1) * 2) * panel_bytes;
                tdm_load_panel(ghi + (size_t)(p + 1) * KP, soff);
                tdm_load_panel(glo + (size_t)(p + 1) * KP, soff + panel_bytes);
                __builtin_amdgcn_s_wait_tensorcnt(2);
            } else {
                __builtin_amdgcn_s_wait_tensorcnt(0);
            }
        }
        __syncthreads();   // panel p visible to all 8 waves

        const bf16_t* bh_base = &sB[p & 1][0][0];
        const bf16_t* bl_base = &sB[p & 1][1][0];

        // B double buffer: load chunk 0, then prefetch c+1 during chunk c
        v16bf Bh[2][4], Bl[2][4];
        #pragma unroll
        for (int nt = 0; nt < 4; ++nt) {
            const int eoff = (nt * 16 + lsub) * KP + half * 16;
            Bh[0][nt] = *(const v16bf*)(bh_base + eoff);
            Bl[0][nt] = *(const v16bf*)(bl_base + eoff);
        }

        #pragma unroll
        for (int c = 0; c < 4; ++c) {
            // ---- prefetch next chunk's B operands from LDS ----
            if (c < 3) {
                #pragma unroll
                for (int nt = 0; nt < 4; ++nt) {
                    const int eoff = (nt * 16 + lsub) * KP + (c + 1) * 32 + half * 16;
                    Bh[(c + 1) & 1][nt] = *(const v16bf*)(bh_base + eoff);
                    Bl[(c + 1) & 1][nt] = *(const v16bf*)(bl_base + eoff);
                }
            }

            // ---- prefetch next A chunk (register double buffer) ----
            const int knext = p * KP + (c + 1) * 32;
            if (knext < H_DIM) {
                v4f* an = a[(c + 1) & 1];
                an[0] = *(const v4f*)(arow + knext);
                an[1] = *(const v4f*)(arow + knext + 4);
                an[2] = *(const v4f*)(arow + knext + 8);
                an[3] = *(const v4f*)(arow + knext + 12);
            }

            // ---- convert current A chunk to bf16 hi/lo ----
            const v4f* ac = a[c & 1];
            float av[16];
            #pragma unroll
            for (int i = 0; i < 4; i++) {
                av[i] = ac[0][i]; av[4 + i] = ac[1][i];
                av[8 + i] = ac[2][i]; av[12 + i] = ac[3][i];
            }
            v16bf ahi, alo;
            #pragma unroll
            for (int i = 0; i < 16; i++) {
                bf16_t h = (bf16_t)av[i];
                ahi[i] = h;
                alo[i] = (bf16_t)(av[i] - (float)h);
            }

            // ---- 4 N-tiles x 3 bf16 WMMAs (ah*bh + al*bh + ah*bl) ----
            const v16bf* bh = Bh[c & 1];
            const v16bf* bl = Bl[c & 1];
            #pragma unroll
            for (int nt = 0; nt < 4; ++nt) {
                acc[nt] = __builtin_amdgcn_wmma_f32_16x16x32_bf16(
                    false, ahi, false, bh[nt], (short)0, acc[nt], false, false);
                acc[nt] = __builtin_amdgcn_wmma_f32_16x16x32_bf16(
                    false, alo, false, bh[nt], (short)0, acc[nt], false, false);
                acc[nt] = __builtin_amdgcn_wmma_f32_16x16x32_bf16(
                    false, ahi, false, bl[nt], (short)0, acc[nt], false, false);
            }
        }
        __syncthreads();   // all waves done with stage (p&1) before its reuse
    }

    // ---- epilogue: C/D layout -> lane L holds (M = r + 8*(L/16), N = L%16) ----
    #pragma unroll
    for (int r = 0; r < 8; r++) {
        const int token = t0 + r + half * 8;
        float m1 = -3.402823466e38f; int i1 = 0;
        float m2 = -3.402823466e38f; int i2 = 0;
        #pragma unroll
        for (int nt = 0; nt < 4; nt++) {
            float v = acc[nt][r];
            logits_out[(size_t)token * N_EXP + nt * 16 + lsub] = v;
            top2_insert(v, nt * 16 + lsub, m1, i1, m2, i2);
        }
        #pragma unroll
        for (int m = 1; m < 16; m <<= 1) {
            float o1 = __shfl_xor(m1, m);
            int  oi1 = __shfl_xor(i1, m);
            float o2 = __shfl_xor(m2, m);
            int  oi2 = __shfl_xor(i2, m);
            top2_insert(o1, oi1, m1, i1, m2, i2);
            top2_insert(o2, oi2, m1, i1, m2, i2);
        }
        if (lsub == 0) {
            // renormalized top-2 weights == softmax over the two top logits
            float e  = __expf(m2 - m1);
            float w1 = 1.0f / (1.0f + e);
            w_out[(size_t)token * 2 + 0]   = w1;
            w_out[(size_t)token * 2 + 1]   = 1.0f - w1;
            idx_out[(size_t)token * 2 + 0] = (float)i1;
            idx_out[(size_t)token * 2 + 1] = (float)i2;
        }
    }
}

// ---------------------------------------------------------------------------
extern "C" void kernel_launch(void* const* d_in, const int* in_sizes, int n_in,
                              void* d_out, int out_size, void* d_ws, size_t ws_size,
                              hipStream_t stream) {
    (void)in_sizes; (void)n_in; (void)out_size; (void)ws_size;
    const float* hidden = (const float*)d_in[0];   // [4,4096,4096] fp32
    const float* gate   = (const float*)d_in[1];   // [64,4096]     fp32

    bf16_t* ghi = (bf16_t*)d_ws;                   // 512 KB
    bf16_t* glo = ghi + (size_t)N_EXP * H_DIM;     // 512 KB

    float* out    = (float*)d_out;
    float* logits = out;                                   // [B,S,64]
    float* wts    = out + (size_t)N_TOK * N_EXP;           // [B,S,2]
    float* idx    = wts + (size_t)N_TOK * 2;               // [B,S,2] (as float)

    gate_split_kernel<<<(N_EXP * H_DIM) / 256, 256, 0, stream>>>(gate, ghi, glo);

    moe_router_kernel<<<N_TOK / 16 / 8, 256, 0, stream>>>(
        hidden, ghi, glo, logits, wts, idx);
}
